// SpatialMultiHeadAttentionMIL_49581102465375
// MI455X (gfx1250) — matrix operand order repaired and checked by reading, not compiled, not run on hardware
//
#include <hip/hip_runtime.h>
#include <hip/hip_fp16.h>

// ---------------------------------------------------------------------------
// Shapes (compile-time constants from the reference)
// ---------------------------------------------------------------------------
#define N_TOK 4096
#define EMB   1024
#define DM    512
#define NHEAD 8
#define HD    64
#define NLAYER 2
#define KMAX  160

typedef __attribute__((ext_vector_type(16))) _Float16 v16h;
typedef __attribute__((ext_vector_type(8)))  _Float16 v8h;
typedef __attribute__((ext_vector_type(8)))  float    v8f;
typedef __attribute__((ext_vector_type(4)))  unsigned u32x4;
typedef __attribute__((ext_vector_type(8)))  unsigned u32x8;

// Epilogue modes
#define EPI_BIAS    0   // out = acc + bias
#define EPI_RESRELU 1   // out = res + relu(acc + bias)
#define EPI_RESADD  2   // out = res + acc + bias
#define EPI_GELU    3   // out = gelu_tanh(acc + bias)

// LDS tile pitch: 40 halves (80B). TDM pad: rows of 64B + 16B pad -> pitch 80B.
#define TPITCH 40
#define TILE_BYTES (64 * TPITCH * 2)   // 5120 B per buffer

// ---------------------------------------------------------------------------
// TDM issue: 2D tile (tile_dim0=32 halves, tile_dim1=64 rows) from a row-major
// f16 tensor [dim1][dim0_stride], with LDS padding to the 40-half pitch.
// ---------------------------------------------------------------------------
__device__ __forceinline__ void tdm_load_tile(unsigned long long ga, unsigned lds,
                                              const u32x8& g1)
{
    u32x4 g0;
    g0[0] = 1u;                                   // count=1, no gather
    g0[1] = lds;                                  // lds_addr (bytes)
    g0[2] = (unsigned)ga;                         // global_addr[31:0]
    g0[3] = (unsigned)(ga >> 32) | 0x80000000u;   // global_addr[56:32] | type=2
    asm volatile("tensor_load_to_lds %0, %1" :: "s"(g0), "s"(g1) : "memory");
}

__device__ __forceinline__ u32x8 tdm_group1(int K, int rows)
{
    // word0: data_size=2B (01b<<16), pad_enable (1<<20),
    //        pad_interval=64B (3<<22), pad_amount=4 DWORDs (3<<25)
    u32x8 g1;
    g1[0] = (1u << 16) | (1u << 20) | (3u << 22) | (3u << 25);
    g1[1] = ((unsigned)K & 0xFFFFu) << 16;                            // tensor_dim0 lo
    g1[2] = ((unsigned)K >> 16) | (((unsigned)rows & 0xFFFFu) << 16); // dim0 hi | dim1 lo
    g1[3] = ((unsigned)rows >> 16) | (32u << 16);                     // dim1 hi | tile_dim0=32
    g1[4] = 64u;                                                      // tile_dim1=64, tile_dim2=0
    g1[5] = (unsigned)K;                                              // tensor_dim0_stride lo
    g1[6] = 0u;
    g1[7] = 0u;
    return g1;
}

// ---------------------------------------------------------------------------
// WMMA GEMM with TDM staging:
//   C[M,Nc] = epilogue(A16[M,K] @ Bt16[Nc,K]^T + bias, res)
// Block = 128 threads (4 waves), 64x64 output tile, double-buffered LDS.
// All descriptor state is computed uniformly (stays in SGPRs); only the
// tensor_load_to_lds issue + TENSORcnt wait are wave-0 guarded.
// ---------------------------------------------------------------------------
template<int MODE, bool W32, bool W16>
__global__ __launch_bounds__(128)
void gemm_wmma_tdm(const _Float16* __restrict__ A16, const _Float16* __restrict__ Bt,
                   const float* __restrict__ bias, const float* __restrict__ res,
                   float* __restrict__ out32, _Float16* __restrict__ out16,
                   int M, int K, int Nc)
{
    __shared__ __align__(16) _Float16 As[2][64 * TPITCH];
    __shared__ __align__(16) _Float16 Bs[2][64 * TPITCH];

    const int tid  = threadIdx.x;
    const int wave = tid >> 5;
    const int lane = tid & 31;
    const int m0   = blockIdx.y * 64;
    const int n0   = blockIdx.x * 64;
    const int fr   = lane & 15;     // fragment row (A: M, B: N)
    const int fh   = lane >> 4;     // fragment half-select
    const int nT   = K >> 5;        // K tiles of 32

    v8f acc[4];
    #pragma unroll
    for (int nb = 0; nb < 4; ++nb)
        #pragma unroll
        for (int i = 0; i < 8; ++i) acc[nb][i] = 0.0f;

    // Uniform descriptor state (computed by every thread -> SGPR-resident)
    const u32x8 g1A = tdm_group1(K, M);
    const u32x8 g1B = tdm_group1(K, Nc);
    unsigned long long gaA = (unsigned long long)(size_t)(A16 + (size_t)m0 * K);
    unsigned long long gaB = (unsigned long long)(size_t)(Bt  + (size_t)n0 * K);
    const unsigned ldsA = (unsigned)(size_t)&As[0][0];
    const unsigned ldsB = (unsigned)(size_t)&Bs[0][0];

    if (wave == 0) {                 // prologue: tile 0 -> buffer 0
        tdm_load_tile(gaA, ldsA, g1A);
        tdm_load_tile(gaB, ldsB, g1B);
    }

    for (int kt = 0; kt < nT; ++kt) {
        const int cur = kt & 1;
        const unsigned boff = (unsigned)(1 - cur) * TILE_BYTES;
        const bool more = (kt + 1 < nT);
        if (wave == 0) {
            if (more) {
                tdm_load_tile(gaA + 64, ldsA + boff, g1A);  // +32 halves in K
                tdm_load_tile(gaB + 64, ldsB + boff, g1B);
                __builtin_amdgcn_s_wait_tensorcnt(2);       // pair for tile kt done
            } else {
                __builtin_amdgcn_s_wait_tensorcnt(0);
            }
        }
        gaA += 64;
        gaB += 64;
        __syncthreads();

        const _Float16* aBase = &As[cur][0];
        const _Float16* bBase = &Bs[cur][0];

        // A fragment: row fr, halves K=[fh*8..+7] and [16+fh*8..+7]
        v16h afrag;
        {
            const v8h lo = *(const v8h*)(aBase + (wave * 16 + fr) * TPITCH + fh * 8);
            const v8h hi = *(const v8h*)(aBase + (wave * 16 + fr) * TPITCH + 16 + fh * 8);
            #pragma unroll
            for (int i = 0; i < 8; ++i) { afrag[i] = lo[i]; afrag[8 + i] = hi[i]; }
        }
        // Load ALL B fragments first (8 ds_load_b128 in flight), then 4 WMMAs
        v16h bfrag[4];
        #pragma unroll
        for (int nb = 0; nb < 4; ++nb) {
            const v8h b0 = *(const v8h*)(bBase + (nb * 16 + fr) * TPITCH + fh * 8);
            const v8h b1 = *(const v8h*)(bBase + (nb * 16 + fr) * TPITCH + 16 + fh * 8);
            #pragma unroll
            for (int i = 0; i < 8; ++i) { bfrag[nb][i] = b0[i]; bfrag[nb][8 + i] = b1[i]; }
        }
        #pragma unroll
        for (int nb = 0; nb < 4; ++nb)
            acc[nb] = __builtin_amdgcn_wmma_f32_16x16x32_f16(
                false, afrag, false, bfrag[nb], (short)0, acc[nb], false, false);
        __syncthreads();
    }

    // Epilogue: C layout -> VGPR i : M = fh*8 + i, N = fr + nb*16
    #pragma unroll
    for (int nb = 0; nb < 4; ++nb) {
        const int n  = n0 + nb * 16 + fr;
        const float bv = bias[n];
        #pragma unroll
        for (int i = 0; i < 8; ++i) {
            const int m = m0 + wave * 16 + fh * 8 + i;
            const size_t o = (size_t)m * Nc + n;
            float v = acc[nb][i] + bv;
            if (MODE == EPI_RESRELU)      v = res[o] + fmaxf(v, 0.0f);
            else if (MODE == EPI_RESADD)  v = res[o] + v;
            else if (MODE == EPI_GELU) {
                const float x = v;
                v = 0.5f * x * (1.0f + tanhf(0.7978845608028654f * (x + 0.044715f * x * x * x)));
            }
            if (W32) out32[o] = v;
            if (W16) out16[o] = (_Float16)v;
        }
    }
}

// ---------------------------------------------------------------------------
// Elementwise f32 -> f16 (4 per thread)
// ---------------------------------------------------------------------------
__global__ __launch_bounds__(256)
void cvt16_k(const float* __restrict__ in, _Float16* __restrict__ out)
{
    const size_t i = ((size_t)blockIdx.x * 256 + threadIdx.x) * 4;
    const float4 f = *(const float4*)(in + i);
    _Float16* o = out + i;
    o[0] = (_Float16)f.x; o[1] = (_Float16)f.y;
    o[2] = (_Float16)f.z; o[3] = (_Float16)f.w;
}

// ---------------------------------------------------------------------------
// Weight transpose + convert: W[K][Nc] f32 -> Wt[Nc][K] f16 (32x32 LDS tiles)
// ---------------------------------------------------------------------------
__global__ __launch_bounds__(256)
void wtrans_k(const float* __restrict__ W, _Float16* __restrict__ Wt, int K, int Nc)
{
    __shared__ float tile[32][33];
    const int t  = threadIdx.x;
    const int k0 = blockIdx.y * 32, n0 = blockIdx.x * 32;
    const int r  = t >> 5, c = t & 31;
    #pragma unroll
    for (int i = 0; i < 32; i += 8)
        tile[r + i][c] = W[(size_t)(k0 + r + i) * Nc + n0 + c];
    __syncthreads();
    #pragma unroll
    for (int i = 0; i < 32; i += 8)
        Wt[(size_t)(n0 + r + i) * K + k0 + c] = (_Float16)tile[c][r + i];
}

// ---------------------------------------------------------------------------
// LayerNorm over last dim (512). One block (256 threads) per row.
// ---------------------------------------------------------------------------
template<typename OT>
__global__ __launch_bounds__(256)
void layernorm_k(const float* __restrict__ in, OT* __restrict__ outp,
                 const float* __restrict__ g, const float* __restrict__ b, float eps)
{
    __shared__ float ssum[256];
    __shared__ float ssq[256];
    const int row = blockIdx.x;
    const int tid = threadIdx.x;
    const float* x = in + (size_t)row * DM;
    const float v0 = x[tid], v1 = x[tid + 256];
    ssum[tid] = v0 + v1;
    ssq[tid]  = v0 * v0 + v1 * v1;
    __syncthreads();
    for (int s = 128; s > 0; s >>= 1) {
        if (tid < s) { ssum[tid] += ssum[tid + s]; ssq[tid] += ssq[tid + s]; }
        __syncthreads();
    }
    const float mean = ssum[0] * (1.0f / DM);
    const float var  = ssq[0] * (1.0f / DM) - mean * mean;
    const float inv  = rsqrtf(var + eps);
    OT* o = outp + (size_t)row * DM;
    o[tid]       = (OT)((v0 - mean) * inv * g[tid]       + b[tid]);
    o[tid + 256] = (OT)((v1 - mean) * inv * g[tid + 256] + b[tid + 256]);
}

// ---------------------------------------------------------------------------
// Sparse gathered attention. One wave per (token n, head hh). f16 output.
// qkv layout: [n][3*DM] with q at col hh*64, k at 512+hh*64, v at 1024+hh*64.
// ---------------------------------------------------------------------------
__global__ __launch_bounds__(128)
void attention_k(const float* __restrict__ qkv, const int* __restrict__ indices,
                 const float* __restrict__ distance, const float* __restrict__ lam_l,
                 _Float16* __restrict__ xo16)
{
    __shared__ float sq[4][64];
    __shared__ float sattn[4][KMAX];
    const int tid  = threadIdx.x;
    const int w    = tid >> 5;
    const int lane = tid & 31;
    const int gw   = blockIdx.x * 4 + w;
    const int n    = gw >> 3;
    const int hh   = gw & 7;
    const int Kh   = (hh == 7) ? 160 : (32 + hh * 16);   // {32,48,...,128,160}

    const float* qrow = qkv + (size_t)n * (3 * DM) + hh * HD;
    const float q0 = qrow[lane * 2], q1 = qrow[lane * 2 + 1];
    sq[w][lane * 2] = q0; sq[w][lane * 2 + 1] = q1;
    float qn = q0 * q0 + q1 * q1;
    #pragma unroll
    for (int off = 16; off > 0; off >>= 1) qn += __shfl_xor(qn, off, 32);
    __syncthreads();

    const float sp = log1pf(expf(lam_l[hh]));   // softplus(lam)
    const int*   idxrow = indices + (size_t)n * KMAX;
    const float* drow   = distance + (size_t)n * N_TOK;

    float lmax = -3.0e38f;
    for (int j = lane; j < Kh; j += 32) {
        const int idx = idxrow[j];
        const float* krow = qkv + (size_t)idx * (3 * DM) + DM + hh * HD;
        float dot = 0.0f, kn = 0.0f;
        #pragma unroll 8
        for (int d = 0; d < HD; ++d) { const float kv = krow[d]; dot += sq[w][d] * kv; kn += kv * kv; }
        float lg = dot * 0.125f - 0.5f * (qn + kn) * 0.125f;    // SCALE = 8
        lg += logf(expf(-sp * drow[idx]) + 1e-6f);
        sattn[w][j] = lg;
        lmax = fmaxf(lmax, lg);
    }
    #pragma unroll
    for (int off = 16; off > 0; off >>= 1) lmax = fmaxf(lmax, __shfl_xor(lmax, off, 32));

    float lsum = 0.0f;
    for (int j = lane; j < Kh; j += 32) {
        const float e = expf(sattn[w][j] - lmax);
        sattn[w][j] = e;
        lsum += e;
    }
    #pragma unroll
    for (int off = 16; off > 0; off >>= 1) lsum += __shfl_xor(lsum, off, 32);
    const float inv = 1.0f / lsum;
    __syncthreads();

    float a0 = 0.0f, a1 = 0.0f;
    for (int j = 0; j < Kh; ++j) {
        const int idx = idxrow[j];
        const float wgt = sattn[w][j];
        const float* vrow = qkv + (size_t)idx * (3 * DM) + 2 * DM + hh * HD;
        a0 += wgt * vrow[lane];
        a1 += wgt * vrow[lane + 32];
    }
    _Float16* orow = xo16 + (size_t)n * DM + hh * HD;
    orow[lane]      = (_Float16)(a0 * inv);
    orow[lane + 32] = (_Float16)(a1 * inv);
}

// ---------------------------------------------------------------------------
// Column mean over 4096 rows -> mean[512]
// ---------------------------------------------------------------------------
__global__ __launch_bounds__(256)
void colmean_k(const float* __restrict__ in, float* __restrict__ mean)
{
    const int c = blockIdx.x * 256 + threadIdx.x;
    float s = 0.0f;
    for (int r = 0; r < N_TOK; ++r) s += in[(size_t)r * DM + c];
    mean[c] = s * (1.0f / N_TOK);
}

// ---------------------------------------------------------------------------
// Head: out[2] = mean[512] @ head_w[512,2] + head_b[2]
// ---------------------------------------------------------------------------
__global__ __launch_bounds__(256)
void head_k(const float* __restrict__ mean, const float* __restrict__ hw,
            const float* __restrict__ hb, float* __restrict__ outp)
{
    __shared__ float s0[256];
    __shared__ float s1[256];
    const int t = threadIdx.x;
    float p0 = 0.0f, p1 = 0.0f;
    for (int d = t; d < DM; d += 256) {
        const float m = mean[d];
        p0 += m * hw[d * 2 + 0];
        p1 += m * hw[d * 2 + 1];
    }
    s0[t] = p0; s1[t] = p1;
    __syncthreads();
    for (int s = 128; s > 0; s >>= 1) {
        if (t < s) { s0[t] += s0[t + s]; s1[t] += s1[t + s]; }
        __syncthreads();
    }
    if (t == 0) { outp[0] = s0[0] + hb[0]; outp[1] = s1[0] + hb[1]; }
}

// ---------------------------------------------------------------------------
// Orchestration
// ---------------------------------------------------------------------------
extern "C" void kernel_launch(void* const* d_in, const int* in_sizes, int n_in,
                              void* d_out, int out_size, void* d_ws, size_t ws_size,
                              hipStream_t stream)
{
    (void)in_sizes; (void)n_in; (void)out_size; (void)ws_size;

    const float* x          = (const float*)d_in[0];
    const float* distance   = (const float*)d_in[1];
    const int*   indices    = (const int*)  d_in[2];
    const float* adapter_w  = (const float*)d_in[5];
    const float* adapter_b  = (const float*)d_in[6];
    const float* res_w      = (const float*)d_in[7];
    const float* res_b      = (const float*)d_in[8];
    const float* ln1_g      = (const float*)d_in[9];
    const float* ln1_b      = (const float*)d_in[10];
    const float* qkv_w      = (const float*)d_in[11];
    const float* qkv_b      = (const float*)d_in[12];
    const float* proj_w     = (const float*)d_in[13];
    const float* proj_b     = (const float*)d_in[14];
    const float* ln2_g      = (const float*)d_in[15];
    const float* ln2_b      = (const float*)d_in[16];
    const float* fc1_w      = (const float*)d_in[17];
    const float* fc1_b      = (const float*)d_in[18];
    const float* fc2_w      = (const float*)d_in[19];
    const float* fc2_b      = (const float*)d_in[20];
    const float* lam        = (const float*)d_in[21];
    const float* lnf_g      = (const float*)d_in[22];
    const float* lnf_b      = (const float*)d_in[23];
    const float* head_w     = (const float*)d_in[24];
    const float* head_b     = (const float*)d_in[25];

    // ---- workspace layout ----
    char* ws = (char*)d_ws;
    const size_t MB = 1ull << 20;
    float*    h     = (float*)(ws + 0 * MB);      // 8 MB residual stream
    float*    hA    = (float*)(ws + 8 * MB);      // 8 MB adapter out / final LN f32
    float*    qkvb  = (float*)(ws + 16 * MB);     // 24 MB
    _Float16* fcb16 = (_Float16*)(ws + 40 * MB);  // 16 MB gelu out (first 8 MB doubles as x16)
    _Float16* x16   = (_Float16*)(ws + 40 * MB);  // alias (consumed before fcb16 written)
    _Float16* xo16  = (_Float16*)(ws + 56 * MB);  // 4 MB attn out (doubles as hA16)
    _Float16* hA16  = (_Float16*)(ws + 56 * MB);  // alias (consumed before xo16 written)
    _Float16* xa16  = (_Float16*)(ws + 60 * MB);  // 4 MB LN f16 out
    _Float16* adT   = (_Float16*)(ws + 64 * MB);  // 1 MB
    _Float16* reT   = (_Float16*)(ws + 65 * MB);  // 0.5 MB
    _Float16* qkT   = (_Float16*)(ws + (65 * MB + 512 * 1024)); // 2 x 1.5 MB
    _Float16* prT   = (_Float16*)(ws + (68 * MB + 512 * 1024)); // 2 x 0.5 MB
    _Float16* f1T   = (_Float16*)(ws + (69 * MB + 512 * 1024)); // 2 x 2 MB
    _Float16* f2T   = (_Float16*)(ws + (73 * MB + 512 * 1024)); // 2 x 2 MB
    float*    cmean = (float*)(ws + 78 * MB);

    const size_t QKT_N = (size_t)DM * 3 * DM;
    const size_t PRT_N = (size_t)DM * DM;
    const size_t F1T_N = (size_t)DM * 4 * DM;
    const size_t F2T_N = (size_t)(4 * DM) * DM;

    // ---- pre-pass: f16 conversions / weight transposes ----
    cvt16_k<<<(N_TOK * EMB) / 1024, 256, 0, stream>>>(x, x16);
    wtrans_k<<<dim3(DM / 32, EMB / 32), 256, 0, stream>>>(adapter_w, adT, EMB, DM);
    wtrans_k<<<dim3(DM / 32, DM / 32),  256, 0, stream>>>(res_w, reT, DM, DM);
    for (int l = 0; l < NLAYER; ++l) {
        wtrans_k<<<dim3(3 * DM / 32, DM / 32), 256, 0, stream>>>(qkv_w + l * QKT_N, qkT + l * QKT_N, DM, 3 * DM);
        wtrans_k<<<dim3(DM / 32, DM / 32),     256, 0, stream>>>(proj_w + l * PRT_N, prT + l * PRT_N, DM, DM);
        wtrans_k<<<dim3(4 * DM / 32, DM / 32), 256, 0, stream>>>(fc1_w + l * F1T_N, f1T + l * F1T_N, DM, 4 * DM);
        wtrans_k<<<dim3(DM / 32, 4 * DM / 32), 256, 0, stream>>>(fc2_w + l * F2T_N, f2T + l * F2T_N, 4 * DM, DM);
    }

    const dim3 blk(128);
    const dim3 gD (DM / 64,     N_TOK / 64);
    const dim3 gQ (3 * DM / 64, N_TOK / 64);
    const dim3 gF (4 * DM / 64, N_TOK / 64);

    // h = x @ adapter_w + b ; h = h + relu(h @ res_w + b)
    gemm_wmma_tdm<EPI_BIAS, true, true><<<gD, blk, 0, stream>>>(
        x16, adT, adapter_b, nullptr, hA, hA16, N_TOK, EMB, DM);
    gemm_wmma_tdm<EPI_RESRELU, true, false><<<gD, blk, 0, stream>>>(
        hA16, reT, res_b, hA, h, nullptr, N_TOK, DM, DM);

    for (int l = 0; l < NLAYER; ++l) {
        layernorm_k<_Float16><<<N_TOK, 256, 0, stream>>>(h, xa16, ln1_g + l * DM, ln1_b + l * DM, 1e-5f);
        gemm_wmma_tdm<EPI_BIAS, true, false><<<gQ, blk, 0, stream>>>(
            xa16, qkT + l * QKT_N, qkv_b + (size_t)l * 3 * DM, nullptr, qkvb, nullptr,
            N_TOK, DM, 3 * DM);
        attention_k<<<(N_TOK * NHEAD) / 4, blk, 0, stream>>>(qkvb, indices, distance,
                                                             lam + l * NHEAD, xo16);
        gemm_wmma_tdm<EPI_RESADD, true, false><<<gD, blk, 0, stream>>>(
            xo16, prT + l * PRT_N, proj_b + (size_t)l * DM, h, h, nullptr,
            N_TOK, DM, DM);
        layernorm_k<_Float16><<<N_TOK, 256, 0, stream>>>(h, xa16, ln2_g + l * DM, ln2_b + l * DM, 1e-5f);
        gemm_wmma_tdm<EPI_GELU, false, true><<<gF, blk, 0, stream>>>(
            xa16, f1T + l * F1T_N, fc1_b + (size_t)l * 4 * DM, nullptr, nullptr, fcb16,
            N_TOK, DM, 4 * DM);
        gemm_wmma_tdm<EPI_RESADD, true, false><<<gD, blk, 0, stream>>>(
            fcb16, f2T + l * F2T_N, fc2_b + (size_t)l * DM, h, h, nullptr,
            N_TOK, 4 * DM, DM);
    }

    layernorm_k<float><<<N_TOK, 256, 0, stream>>>(h, hA, lnf_g, lnf_b, 1e-6f);
    colmean_k<<<DM / 256, 256, 0, stream>>>(hA, cmean);
    head_k<<<1, 256, 0, stream>>>(cmean, head_w, head_b, (float*)d_out);
}
